// LightweightConvolution2D_23347442221640
// MI455X (gfx1250) — matrix operand order repaired and checked by reading, not compiled, not run on hardware
//
#include <hip/hip_runtime.h>
#include <math.h>

// Problem constants (from the reference)
#define Bdim 16
#define Tdim 1024
#define Cdim 1024
#define Hdim 16
#define Kw   31
#define PADw 15
#define Mdim (Bdim * Tdim)   // 16384 rows

typedef __attribute__((ext_vector_type(4)))  __bf16 v4bf;
typedef __attribute__((ext_vector_type(8)))  __bf16 v8bf;
typedef __attribute__((ext_vector_type(16))) __bf16 v16bf;
typedef __attribute__((ext_vector_type(8)))  float  v8f;
typedef int v4i_ __attribute__((vector_size(16)));   // matches builtin param

// GEMM tiling: 256 threads = 8 waves; block tile 128(M) x 64(N); waves 4x2,
// each wave owns 32x32 = 2x2 WMMA accums.  KT == bf16 WMMA K (32).
#define TM 128
#define TN 64
#define KT 32

// ---------------- async global->LDS copy (CDNA5), with fallback -------------
#if __has_builtin(__builtin_amdgcn_global_load_async_to_lds_b128)
#define HAVE_ASYNC_LDS 1
#endif

__device__ __forceinline__ void async_copy16(const void* g, void* l) {
#ifdef HAVE_ASYNC_LDS
  // builtin signature (from clang diagnostic): (v4i AS1*, v4i AS3*, imm, imm)
  __attribute__((address_space(1))) void* g1 =
      (__attribute__((address_space(1))) void*)(void*)(size_t)(const char*)g;
  __attribute__((address_space(3))) void* l3 =
      (__attribute__((address_space(3))) void*)l;
  __builtin_amdgcn_global_load_async_to_lds_b128(
      (__attribute__((address_space(1))) v4i_*)g1,
      (__attribute__((address_space(3))) v4i_*)l3, 0, 0);
#else
  *(float4*)l = *(const float4*)g;
#endif
}

template <int N>
__device__ __forceinline__ void wait_async() {
#ifdef HAVE_ASYNC_LDS
#if __has_builtin(__builtin_amdgcn_s_wait_asynccnt)
  __builtin_amdgcn_s_wait_asynccnt((unsigned short)N);
#else
  asm volatile("s_wait_asynccnt %0" ::"i"(N));
#endif
#endif
}

// ---------------- bf16 split-precision helpers (bf16x3 fp32 GEMM) -----------
__device__ __forceinline__ void split4(const float4 v, v4bf* hi, v4bf* lo) {
  v4bf h, l;
  h[0] = (__bf16)v.x; l[0] = (__bf16)(v.x - (float)h[0]);
  h[1] = (__bf16)v.y; l[1] = (__bf16)(v.y - (float)h[1]);
  h[2] = (__bf16)v.z; l[2] = (__bf16)(v.z - (float)h[2]);
  h[3] = (__bf16)v.w; l[3] = (__bf16)(v.w - (float)h[3]);
  *hi = h; *lo = l;
}

// A fragment, 16x32 bf16 (ISA 7.12.2): lane half=lane>>4, row=lane&15.
// elements 0..7  -> K in [8*half, 8*half+8); 8..15 -> K in [16+8*half, +8)
__device__ __forceinline__ v16bf frag_a(const __bf16* rowbase, int half) {
  v8bf p0 = *(const v8bf*)(rowbase + 8 * half);
  v8bf p1 = *(const v8bf*)(rowbase + 16 + 8 * half);
  return __builtin_shufflevector(p0, p1, 0, 1, 2, 3, 4, 5, 6, 7,
                                         8, 9, 10, 11, 12, 13, 14, 15);
}
// B fragment, 32x16 bf16: N = lane&15; elements 0..7 -> K in [8*half, +8), 8..15 -> K in [16+8*half, +8)
__device__ __forceinline__ v16bf frag_b(const __bf16* colbase, int half) {
  v8bf p0 = *(const v8bf*)(colbase + 8 * half);
  v8bf p1 = *(const v8bf*)(colbase + 16 + 8 * half);
  return __builtin_shufflevector(p0, p1, 0, 1, 2, 3, 4, 5, 6, 7,
                                         8, 9, 10, 11, 12, 13, 14, 15);
}

// acc += A*B with A=ah+al, B=bh+bl (drop al*bl, O(2^-16) relative)
__device__ __forceinline__ v8f mma3(v16bf ah, v16bf al, v16bf bh, v16bf bl,
                                    v8f acc) {
  acc = __builtin_amdgcn_wmma_f32_16x16x32_bf16(false, al, false, bh, (short)0,
                                                acc, false, false);
  acc = __builtin_amdgcn_wmma_f32_16x16x32_bf16(false, ah, false, bl, (short)0,
                                                acc, false, false);
  acc = __builtin_amdgcn_wmma_f32_16x16x32_bf16(false, ah, false, bh, (short)0,
                                                acc, false, false);
  return acc;
}

// ---------------------------------------------------------------------------
// Kernel 0: pack fp32 -> bf16 hi/lo planes (one pass; removes all per-tile
// conversion work from the GEMM hot loops).
// ---------------------------------------------------------------------------
__global__ __launch_bounds__(256) void pack_kernel(
    const float* __restrict__ src, __bf16* __restrict__ hi,
    __bf16* __restrict__ lo, int n4) {
  const int i = blockIdx.x * 256 + threadIdx.x;
  if (i < n4) {
    float4 v = ((const float4*)src)[i];
    v4bf h, l; split4(v, &h, &l);
    ((v4bf*)hi)[i] = h;
    ((v4bf*)lo)[i] = l;
  }
}

// ---------------- double-buffered LDS tile structs --------------------------
struct G1Tiles {
  __bf16 Ah[TM][KT];  __bf16 Al[TM][KT];
  __bf16 Bah[TN][KT]; __bf16 Bal[TN][KT];
  __bf16 Bgh[TN][KT]; __bf16 Bgl[TN][KT];
};  // 32 KB
struct G2Tiles {
  __bf16 Ah[TM][KT]; __bf16 Al[TM][KT];
  __bf16 Bh[TN][KT]; __bf16 Bl[TN][KT];
};  // 24 KB

// ---------------------------------------------------------------------------
// Kernel 1: X = GLU(query @ W1^T + b1) -> (M, C), bf16x3 WMMA, async staging.
// Per-thread async issue per stage: A 2x2 + B 1x4 = 8 chunks (uniform).
// ---------------------------------------------------------------------------
__global__ __launch_bounds__(256) void gemm1_glu_kernel(
    const __bf16* __restrict__ Qh, const __bf16* __restrict__ Ql,
    const __bf16* __restrict__ W1h, const __bf16* __restrict__ W1l,
    const float* __restrict__ b1, float* __restrict__ X) {
  extern __shared__ __align__(32) char smem_raw[];
  G1Tiles* bufs = (G1Tiles*)smem_raw;   // [2]

  const int tid  = threadIdx.x;
  const int lane = tid & 31;
  const int wave = tid >> 5;
  const int wm   = wave & 3;
  const int wn   = wave >> 2;
  const int gm0  = blockIdx.x * TM;
  const int gn0  = blockIdx.y * TN;
  const int half = lane >> 4;
  const int lid  = lane & 15;

  v8f acc_a[2][2] = {};
  v8f acc_g[2][2] = {};

  auto issue = [&](int k0, G1Tiles* tb) {
    for (int s = tid; s < TM * 4; s += 256) {       // 512 chunks/plane, 2 iters
      const int r = s >> 2, seg = (s & 3) * 8;      // 8 bf16 per 16B chunk
      const size_t off = (size_t)(gm0 + r) * Cdim + k0 + seg;
      async_copy16(Qh + off, &tb->Ah[r][seg]);
      async_copy16(Ql + off, &tb->Al[r][seg]);
    }
    for (int s = tid; s < TN * 4; s += 256) {       // 256 chunks/plane, 1 iter
      const int r = s >> 2, seg = (s & 3) * 8;
      const size_t offA = (size_t)(gn0 + r) * Cdim + k0 + seg;
      const size_t offG = (size_t)(gn0 + r + Cdim) * Cdim + k0 + seg;
      async_copy16(W1h + offA, &tb->Bah[r][seg]);
      async_copy16(W1l + offA, &tb->Bal[r][seg]);
      async_copy16(W1h + offG, &tb->Bgh[r][seg]);
      async_copy16(W1l + offG, &tb->Bgl[r][seg]);
    }
  };

  auto compute = [&](G1Tiles* tb) {
    v16bf ah[2], al[2], bah[2], bal[2], bgh[2], bgl[2];
    for (int i = 0; i < 2; ++i) {
      const int arow = wm * 32 + i * 16 + lid;
      ah[i] = frag_a(&tb->Ah[arow][0], half);
      al[i] = frag_a(&tb->Al[arow][0], half);
      const int bcol = wn * 32 + i * 16 + lid;
      bah[i] = frag_b(&tb->Bah[bcol][0], half);
      bal[i] = frag_b(&tb->Bal[bcol][0], half);
      bgh[i] = frag_b(&tb->Bgh[bcol][0], half);
      bgl[i] = frag_b(&tb->Bgl[bcol][0], half);
    }
    for (int i = 0; i < 2; ++i)
      for (int j = 0; j < 2; ++j) {
        acc_a[i][j] = mma3(ah[i], al[i], bah[j], bal[j], acc_a[i][j]);
        acc_g[i][j] = mma3(ah[i], al[i], bgh[j], bgl[j], acc_g[i][j]);
      }
  };

  issue(0, &bufs[0]);
  int p = 0;
  for (int k0 = KT; k0 < Cdim; k0 += KT) {
    issue(k0, &bufs[p ^ 1]);
    wait_async<8>();          // current stage landed; next stage in flight
    __syncthreads();
    compute(&bufs[p]);
    __syncthreads();
    p ^= 1;
  }
  wait_async<0>();
  __syncthreads();
  compute(&bufs[p]);

  // Epilogue: GLU with bias.  C/D: VGPR r -> row r / r+8, col = lane&15.
  for (int i = 0; i < 2; ++i)
    for (int j = 0; j < 2; ++j) {
      const int n = gn0 + wn * 32 + j * 16 + lid;
      const float bia = b1[n];
      const float big = b1[n + Cdim];
      for (int r = 0; r < 8; ++r) {
        const int m = gm0 + wm * 32 + i * 16 + r + 8 * half;
        const float va = acc_a[i][j][r] + bia;
        const float vg = acc_g[i][j][r] + big;
        X[(size_t)m * Cdim + n] = va * (1.0f / (1.0f + expf(-vg)));
      }
    }
}

// ---------------------------------------------------------------------------
// Kernel 2: convs + mask + concat, emitting Y directly as bf16 hi/lo planes.
// One block per (b,t); X (64MB) is L2-resident so the 31x strided re-reads
// for the time conv are L2-served.
// ---------------------------------------------------------------------------
__global__ __launch_bounds__(256) void conv_concat_kernel(
    const float* __restrict__ X,    // (M, C)
    const float* __restrict__ Wt,   // (H,1,K)
    const float* __restrict__ Wf,   // (1,1,K)
    const int*   __restrict__ mask, // (B,1,T)
    __bf16* __restrict__ Yh,        // (M, 2C) hi plane [xc | xf]
    __bf16* __restrict__ Yl)        // (M, 2C) lo plane
{
  __shared__ float sw[Hdim + 1][Kw];
  __shared__ float rowbuf[Cdim + 2 * PADw];

  const int tid = threadIdx.x;
  const int bt  = blockIdx.x;
  const int b   = bt / Tdim;
  const int t   = bt % Tdim;

  if (tid < Hdim + 1) {
    const float* src = (tid < Hdim) ? (Wt + tid * Kw) : Wf;
    float mx = src[0];
    for (int j = 1; j < Kw; ++j) mx = fmaxf(mx, src[j]);
    float e[Kw];
    float s = 0.0f;
    for (int j = 0; j < Kw; ++j) { e[j] = expf(src[j] - mx); s += e[j]; }
    const float inv = 1.0f / s;
    for (int j = 0; j < Kw; ++j) sw[tid][j] = e[j] * inv;
  }
  for (int i = tid; i < Cdim + 2 * PADw; i += 256) {
    const int c = i - PADw;
    rowbuf[i] = (c >= 0 && c < Cdim) ? X[(size_t)bt * Cdim + c] : 0.0f;
  }
  __syncthreads();

  const int msk = mask[b * Tdim + t];

  for (int c = tid; c < Cdim; c += 256) {
    float xf = 0.0f;
    for (int j = 0; j < Kw; ++j) xf += sw[Hdim][j] * rowbuf[c + j];

    const float* wrow = sw[c & (Hdim - 1)];
    float xt = 0.0f;
    for (int j = 0; j < Kw; ++j) {
      const int tt = t + j - PADw;
      if (tt >= 0 && tt < Tdim)
        xt += wrow[j] * X[((size_t)b * Tdim + tt) * Cdim + c];
    }
    if (msk == 0) { xt = 0.0f; xf = 0.0f; }

    const size_t i0 = (size_t)bt * (2 * Cdim) + c;          // xc half
    const size_t i1 = i0 + Cdim;                            // xf half
    __bf16 h0 = (__bf16)xt; Yh[i0] = h0; Yl[i0] = (__bf16)(xt - (float)h0);
    __bf16 h1 = (__bf16)xf; Yh[i1] = h1; Yl[i1] = (__bf16)(xf - (float)h1);
  }
}

// ---------------------------------------------------------------------------
// Kernel 3: Out = Y @ W2^T + b2   (M,2C)@(2C,C) -> (M,C), async staging.
// Per-thread async issue per stage: A 2x2 + B 1x2 = 6 chunks (uniform).
// ---------------------------------------------------------------------------
__global__ __launch_bounds__(256) void gemm2_kernel(
    const __bf16* __restrict__ Yh, const __bf16* __restrict__ Yl,
    const __bf16* __restrict__ W2h, const __bf16* __restrict__ W2l,
    const float* __restrict__ b2, float* __restrict__ Out) {
  extern __shared__ __align__(32) char smem_raw[];
  G2Tiles* bufs = (G2Tiles*)smem_raw;   // [2]

  const int tid  = threadIdx.x;
  const int lane = tid & 31;
  const int wave = tid >> 5;
  const int wm   = wave & 3;
  const int wn   = wave >> 2;
  const int gm0  = blockIdx.x * TM;
  const int gn0  = blockIdx.y * TN;
  const int half = lane >> 4;
  const int lid  = lane & 15;
  const int K2   = 2 * Cdim;

  v8f acc[2][2] = {};

  auto issue = [&](int k0, G2Tiles* tb) {
    for (int s = tid; s < TM * 4; s += 256) {
      const int r = s >> 2, seg = (s & 3) * 8;
      const size_t off = (size_t)(gm0 + r) * K2 + k0 + seg;
      async_copy16(Yh + off, &tb->Ah[r][seg]);
      async_copy16(Yl + off, &tb->Al[r][seg]);
    }
    for (int s = tid; s < TN * 4; s += 256) {
      const int r = s >> 2, seg = (s & 3) * 8;
      const size_t off = (size_t)(gn0 + r) * K2 + k0 + seg;
      async_copy16(W2h + off, &tb->Bh[r][seg]);
      async_copy16(W2l + off, &tb->Bl[r][seg]);
    }
  };

  auto compute = [&](G2Tiles* tb) {
    v16bf ah[2], al[2], bh[2], bl[2];
    for (int i = 0; i < 2; ++i) {
      const int arow = wm * 32 + i * 16 + lid;
      ah[i] = frag_a(&tb->Ah[arow][0], half);
      al[i] = frag_a(&tb->Al[arow][0], half);
      const int bcol = wn * 32 + i * 16 + lid;
      bh[i] = frag_b(&tb->Bh[bcol][0], half);
      bl[i] = frag_b(&tb->Bl[bcol][0], half);
    }
    for (int i = 0; i < 2; ++i)
      for (int j = 0; j < 2; ++j)
        acc[i][j] = mma3(ah[i], al[i], bh[j], bl[j], acc[i][j]);
  };

  issue(0, &bufs[0]);
  int p = 0;
  for (int k0 = KT; k0 < K2; k0 += KT) {
    issue(k0, &bufs[p ^ 1]);
    wait_async<6>();
    __syncthreads();
    compute(&bufs[p]);
    __syncthreads();
    p ^= 1;
  }
  wait_async<0>();
  __syncthreads();
  compute(&bufs[p]);

  for (int i = 0; i < 2; ++i)
    for (int j = 0; j < 2; ++j) {
      const int n  = gn0 + wn * 32 + j * 16 + lid;
      const float bn = b2[n];
      for (int r = 0; r < 8; ++r) {
        const int m = gm0 + wm * 32 + i * 16 + r + 8 * half;
        Out[(size_t)m * Cdim + n] = acc[i][j][r] + bn;
      }
    }
}

// ---------------------------------------------------------------------------
// Workspace layout (bytes), with aliasing (regions reused once dead):
//   [  0,  32M) Qh      -> later reused as Yh[0..)
//   [ 32M, 64M) Ql
//   [ 64M, 68M) W1h     -> later reused as Yl[0..)
//   [ 68M, 72M) W1l
//   [  0,  64M) Yh   (aliases Qh/Ql; written by conv after gemm1)
//   [ 64M,128M) Yl   (aliases W1 planes)
//   [128M,192M) X    (fp32 GLU output)
//   [192M,196M) W2h
//   [196M,200M) W2l
// Total 200 MB; every region fully rewritten each call -> deterministic.
// ---------------------------------------------------------------------------
extern "C" void kernel_launch(void* const* d_in, const int* in_sizes, int n_in,
                              void* d_out, int out_size, void* d_ws, size_t ws_size,
                              hipStream_t stream) {
  (void)in_sizes; (void)n_in; (void)out_size; (void)ws_size;

  const float* q    = (const float*)d_in[0];
  const int*   mask = (const int*)  d_in[3];
  const float* W1   = (const float*)d_in[4];
  const float* b1   = (const float*)d_in[5];
  const float* W2   = (const float*)d_in[6];
  const float* b2   = (const float*)d_in[7];
  const float* wt   = (const float*)d_in[8];
  const float* wf   = (const float*)d_in[9];

  const size_t MB = 1024ull * 1024ull;
  char* ws = (char*)d_ws;
  __bf16* Qh  = (__bf16*)(ws + 0 * MB);
  __bf16* Ql  = (__bf16*)(ws + 32 * MB);
  __bf16* W1h = (__bf16*)(ws + 64 * MB);
  __bf16* W1l = (__bf16*)(ws + 68 * MB);
  __bf16* Yh  = (__bf16*)(ws + 0 * MB);     // aliases Qh/Ql (dead after gemm1)
  __bf16* Yl  = (__bf16*)(ws + 64 * MB);    // aliases W1 planes
  float*  X   = (float*) (ws + 128 * MB);
  __bf16* W2h = (__bf16*)(ws + 192 * MB);
  __bf16* W2l = (__bf16*)(ws + 196 * MB);
  float*  Out = (float*)d_out;

  // one-pass fp32 -> bf16 hi/lo packing
  const int n4q = Mdim * Cdim / 4;          // 4,194,304
  const int n4w = 2 * Cdim * Cdim / 4;      //   524,288
  pack_kernel<<<n4q / 256, 256, 0, stream>>>(q,  Qh,  Ql,  n4q);
  pack_kernel<<<n4w / 256, 256, 0, stream>>>(W1, W1h, W1l, n4w);
  pack_kernel<<<n4w / 256, 256, 0, stream>>>(W2, W2h, W2l, n4w);

  dim3 gemmGrid(Mdim / TM, Cdim / TN);      // (128, 16)
  gemm1_glu_kernel<<<gemmGrid, 256, 2 * sizeof(G1Tiles), stream>>>(
      Qh, Ql, W1h, W1l, b1, X);
  conv_concat_kernel<<<Mdim, 256, 0, stream>>>(X, wt, wf, mask, Yh, Yl);
  gemm2_kernel<<<gemmGrid, 256, 2 * sizeof(G2Tiles), stream>>>(
      Yh, Yl, W2h, W2l, b2, Out);
}
